// Generator_8203387535557
// MI455X (gfx1250) — matrix operand order repaired and checked by reading, not compiled, not run on hardware
//
#include <hip/hip_runtime.h>

// ---------------------------------------------------------------------------
// Fused 3-layer MLP (384 -> 256 -> 512 -> 1152) over 32768 tokens, gfx1250.
// bf16 WMMA (v_wmma_f32_16x16x32_bf16), f32 accumulate, intermediates in LDS.
// ---------------------------------------------------------------------------

typedef __attribute__((ext_vector_type(16))) __bf16 v16bf;
typedef __attribute__((ext_vector_type(8)))  float  v8f;

#define LEAKY_ALPHA 0.2f

constexpr int Bn   = 64;
constexpr int S    = 512;
constexpr int NTOK = Bn * S;          // 32768 tokens
constexpr int CD   = 256;             // content dim
constexpr int MD   = 128;             // motion dim
constexpr int D0   = CD + MD;         // 384
constexpr int D1   = 256;
constexpr int D2   = 512;
constexpr int D3   = 1152;

constexpr int RT    = 64;             // row (token) tile per workgroup
constexpr int BLOCK = 256;            // 8 wave32
constexpr int NWAVE = BLOCK / 32;
constexpr int MI    = RT / 16;        // 4 row sub-tiles per column strip

// Padded LDS strides (elements). Chosen so A-fragment reads land on distinct
// banks: stride/2 dwords % 64 == 4 -> lane M hits bank 4*M.
constexpr int LDA0 = D0 + 8;          // 392
constexpr int LDH1 = D1 + 8;          // 264
constexpr int LDH2 = D2 + 8;          // 520

constexpr int REGION1    = RT * LDH2 * 2;             // 66560 B: sA (50176 B) then sH2 alias
constexpr int SMEM_BYTES = REGION1 + RT * LDH1 * 2;   // + 33792 B = 100352 B (~98 KB)

static_assert(sizeof(__bf16) == 2, "bf16 must be 2 bytes");

__device__ __forceinline__ float leaky(float x) {
    return x >= 0.0f ? x : LEAKY_ALPHA * x;
}

// One GEMM layer for this workgroup's RT x KDIM activation tile (bf16 in LDS)
// against f32 weights W [KDIM x NOUT] streamed from L2 (converted to bf16 on
// the fly). Output: bias (+ optional leaky) -> bf16 LDS tile, or f32 global.
template <int KDIM, int NOUT, bool ACT, bool TO_LDS>
__device__ __forceinline__ void mlp_layer(
    const __bf16* __restrict__ sA, int lda,
    const float*  __restrict__ W,  const float* __restrict__ bias,
    __bf16* __restrict__ sOut, int ldo,
    float*  __restrict__ gOut, int rowBase,
    int lane, int wave)
{
    const int Nl   = lane & 15;   // column within 16-wide strip / A-row M
    const int half = lane >> 4;   // lane group (0: lanes 0-15, 1: lanes 16-31)

    for (int ni = wave; ni < NOUT / 16; ni += NWAVE) {
        const int colBase = ni * 16;

        v8f acc[MI];
        const v8f vzero = {0.f, 0.f, 0.f, 0.f, 0.f, 0.f, 0.f, 0.f};
#pragma unroll
        for (int mi = 0; mi < MI; ++mi) acc[mi] = vzero;

        for (int kk = 0; kk < KDIM; kk += 32) {
            // ---- B fragment (32x16): VGPR v holds K = (half*16 + 2v, +1), col Nl
            v16bf bf;
            const float* pw = W + (size_t)(kk + (half << 4)) * NOUT + colBase + Nl;
#pragma unroll
            for (int v = 0; v < 8; ++v) {
                bf[2 * v]     = (__bf16)pw[(size_t)(2 * v) * NOUT];
                bf[2 * v + 1] = (__bf16)pw[(size_t)(2 * v + 1) * NOUT];
            }
            if (!TO_LDS) {
                // prefetch next K-block of the (large) W3 into GL2/L0
                __builtin_prefetch(pw + (size_t)32 * NOUT, 0, 0);
            }

            // ---- 4 row sub-tiles reuse this B fragment
#pragma unroll
            for (int mi = 0; mi < MI; ++mi) {
                v16bf af;
                const __bf16* pa = sA + (size_t)(mi * 16 + Nl) * lda + kk;
#pragma unroll
                for (int v = 0; v < 8; ++v) {
                    // A uses contiguous k order: K = 16*half + 2*v (matches B)
                    const int k0 = (half << 4) + (v << 1);
                    af[2 * v]     = pa[k0];
                    af[2 * v + 1] = pa[k0 + 1];
                }
                acc[mi] = __builtin_amdgcn_wmma_f32_16x16x32_bf16(
                    false, af, false, bf, (short)0, acc[mi], false, false);
            }
        }

        // ---- epilogue: bias (+ activation), write out
        const float bv = bias[colBase + Nl];
#pragma unroll
        for (int mi = 0; mi < MI; ++mi) {
#pragma unroll
            for (int v = 0; v < 8; ++v) {
                const int M = mi * 16 + v + (half << 3);  // C/D layout row
                float x = acc[mi][v] + bv;
                if (ACT) x = leaky(x);
                if (TO_LDS)
                    sOut[(size_t)M * ldo + colBase + Nl] = (__bf16)x;
                else
                    gOut[(size_t)(rowBase + M) * D3 + colBase + Nl] = x;
            }
        }
    }
}

__global__ void __launch_bounds__(BLOCK)
fused_mlp_kernel(const float* __restrict__ content,
                 const float* __restrict__ motion,
                 const float* __restrict__ W1, const float* __restrict__ b1,
                 const float* __restrict__ W2, const float* __restrict__ b2,
                 const float* __restrict__ W3, const float* __restrict__ b3,
                 float* __restrict__ out)
{
    extern __shared__ char smem[];
    __bf16* sA  = (__bf16*)smem;              // RT x LDA0 (dead after layer 1)
    __bf16* sH2 = (__bf16*)smem;              // RT x LDH2 (aliases sA)
    __bf16* sH1 = (__bf16*)(smem + REGION1);  // RT x LDH1

    const int tid     = threadIdx.x;
    const int lane    = tid & 31;
    const int wave    = tid >> 5;
    const int rowBase = blockIdx.x * RT;

    // ---- Phase 0: stage X = concat(content[b], motion[b,s]) tile as bf16
    for (int idx = tid; idx < RT * D0; idx += BLOCK) {
        const int r = idx / D0;
        const int c = idx - r * D0;
        const int g = rowBase + r;                      // flat token = b*S + s
        const float x = (c < CD)
            ? content[(size_t)(g / S) * CD + c]
            : motion[(size_t)g * MD + (c - CD)];
        sA[(size_t)r * LDA0 + c] = (__bf16)x;
    }
    __syncthreads();

    // ---- Layer 1: H1 = leaky(X @ W1 + b1)   [RT x 256] -> LDS bf16
    mlp_layer<D0, D1, true, true>(sA, LDA0, W1, b1, sH1, LDH1, nullptr, 0, lane, wave);
    __syncthreads();

    // ---- Layer 2: H2 = leaky(H1 @ W2 + b2)  [RT x 512] -> LDS bf16 (aliases sA)
    mlp_layer<D1, D2, true, true>(sH1, LDH1, W2, b2, sH2, LDH2, nullptr, 0, lane, wave);
    __syncthreads();

    // ---- Layer 3: OUT = H2 @ W3 + b3        [RT x 1152] -> global f32
    mlp_layer<D2, D3, false, false>(sH2, LDH2, W3, b3, nullptr, 0, out, rowBase, lane, wave);
}

extern "C" void kernel_launch(void* const* d_in, const int* in_sizes, int n_in,
                              void* d_out, int out_size, void* d_ws, size_t ws_size,
                              hipStream_t stream)
{
    const float* content = (const float*)d_in[0];
    const float* motion  = (const float*)d_in[1];
    const float* W1      = (const float*)d_in[2];
    const float* b1      = (const float*)d_in[3];
    const float* W2      = (const float*)d_in[4];
    const float* b2      = (const float*)d_in[5];
    const float* W3      = (const float*)d_in[6];
    const float* b3      = (const float*)d_in[7];
    float*       out     = (float*)d_out;

    (void)in_sizes; (void)n_in; (void)out_size; (void)d_ws; (void)ws_size;

    // ~98 KB dynamic LDS per workgroup (gfx1250 WGP has 320 KB).
    (void)hipFuncSetAttribute((const void*)fused_mlp_kernel,
                              hipFuncAttributeMaxDynamicSharedMemorySize,
                              (int)SMEM_BYTES);

    fused_mlp_kernel<<<NTOK / RT, BLOCK, SMEM_BYTES, stream>>>(
        content, motion, W1, b1, W2, b2, W3, b3, out);
}
